// Graph_module_net_0_18631568130110
// MI455X (gfx1250) — compile-verified, matrix operand order checked by reading, not run on hardware
//
#include <hip/hip_runtime.h>
#include <hip/hip_bf16.h>

typedef __attribute__((ext_vector_type(16))) _Float16 v16h;
typedef __attribute__((ext_vector_type(8)))  _Float16 v8h;
typedef __attribute__((ext_vector_type(8)))  float    v8f;

constexpr int Bn = 8;
constexpr int Nn = 1024;
constexpr int Cn = 256;
constexpr int Hn = 4;
constexpr int Kt = 16;             // node_num (top-k)
constexpr int WROW = 2 * Cn + 4;   // 516, row length of W1/W2

// ---------------------------------------------------------------------------
// WMMA helpers (CDNA5, wave32, 16x16x32 f16 -> f32)
// ---------------------------------------------------------------------------
__device__ __forceinline__ v8f wmma16(v16h a, v16h b, v8f c) {
  return __builtin_amdgcn_wmma_f32_16x16x32_f16(
      /*neg_a=*/false, a, /*neg_b=*/false, b,
      /*c_mod=*/(short)0, c, /*reuse_a=*/false, /*reuse_b=*/false);
}

// Load a 16x32 f16 fragment from a row-major [row][k] array (A operand, or the
// rows of B^T for the B operand).  Per ISA 7.12.2: lane L -> row = L&15,
// halves kb..kb+7 and kb+16..kb+23 with kb = k0 + ((L>>4)<<3).
__device__ __forceinline__ v16h load_frag(const _Float16* __restrict__ base,
                                          int ldk, int k0) {
  int lane = threadIdx.x & 31;
  int row  = lane & 15;
  int kb   = k0 + ((lane >> 4) << 3);
  const _Float16* p = base + (size_t)row * ldk + kb;
  union { v16h v; v8h h[2]; } u;
  u.h[0] = *(const v8h*)(p);
  u.h[1] = *(const v8h*)(p + 16);
  return u.v;
}

__device__ __forceinline__ float wred(float v) {
#pragma unroll
  for (int m = 16; m >= 1; m >>= 1) v += __shfl_xor(v, m, 32);
  return v;
}

// ---------------------------------------------------------------------------
// K_misc: weights -> f16, zero the shared `present` vectors
// ---------------------------------------------------------------------------
__global__ void misc_kernel(const float* __restrict__ cw1,
                            const float* __restrict__ cw2,
                            _Float16* __restrict__ w1h,
                            _Float16* __restrict__ w2h,
                            float* __restrict__ p1, float* __restrict__ p2) {
  for (int i = threadIdx.x; i < Hn * 64 * 64; i += blockDim.x) {
    w1h[i] = (_Float16)cw1[i];
    w2h[i] = (_Float16)cw2[i];
  }
  for (int i = threadIdx.x; i < Nn; i += blockDim.x) { p1[i] = 0.f; p2[i] = 0.f; }
}

// ---------------------------------------------------------------------------
// prep1: per row (b,n): f16 copy, normalized f16 copy, qt/kt (H=4 each)
// 256 threads = 8 waves, one wave per row.
// ---------------------------------------------------------------------------
__global__ __launch_bounds__(256) void prep1_kernel(
    const float* __restrict__ x, const float* __restrict__ spat,
    const float* __restrict__ W, _Float16* __restrict__ xh,
    _Float16* __restrict__ xnh, float* __restrict__ qt, float* __restrict__ kt) {
  int wave = threadIdx.x >> 5, lane = threadIdx.x & 31;
  int rowg = blockIdx.x * 8 + wave;                 // b*N + n
  const float* xr = x + (size_t)rowg * Cn;
  float xv[8], ss = 0.f, qa[4] = {0,0,0,0}, ka[4] = {0,0,0,0};
#pragma unroll
  for (int t = 0; t < 8; ++t) {
    int c = t * 32 + lane;
    float v = xr[c];
    xv[t] = v; ss += v * v;
#pragma unroll
    for (int h = 0; h < 4; ++h) {
      qa[h] += v * W[h * WROW + c];
      ka[h] += v * W[h * WROW + Cn + c];
    }
  }
  ss = wred(ss);
#pragma unroll
  for (int h = 0; h < 4; ++h) { qa[h] = wred(qa[h]); ka[h] = wred(ka[h]); }
  float sc = 1.f / fmaxf(sqrtf(ss), 1e-8f);
#pragma unroll
  for (int t = 0; t < 8; ++t) {
    int c = t * 32 + lane;
    xh[(size_t)rowg * Cn + c]  = (_Float16)xv[t];
    xnh[(size_t)rowg * Cn + c] = (_Float16)(xv[t] * sc);
  }
  if (lane < 4) {
    int h = lane;
    float s0 = spat[rowg * 2], s1 = spat[rowg * 2 + 1];
    qt[rowg * 4 + h] = qa[h] + s0 * W[h * WROW + 2*Cn]     + s1 * W[h * WROW + 2*Cn + 1];
    kt[rowg * 4 + h] = ka[h] + s0 * W[h * WROW + 2*Cn + 2] + s1 * W[h * WROW + 2*Cn + 3];
  }
}

// prep2: same but input rows are f16 (out1 transposed), no raw f16 copy
__global__ __launch_bounds__(256) void prep2_kernel(
    const _Float16* __restrict__ x, const float* __restrict__ spat,
    const float* __restrict__ W, _Float16* __restrict__ xnh,
    float* __restrict__ qt, float* __restrict__ kt) {
  int wave = threadIdx.x >> 5, lane = threadIdx.x & 31;
  int rowg = blockIdx.x * 8 + wave;
  const _Float16* xr = x + (size_t)rowg * Cn;
  float xv[8], ss = 0.f, qa[4] = {0,0,0,0}, ka[4] = {0,0,0,0};
#pragma unroll
  for (int t = 0; t < 8; ++t) {
    int c = t * 32 + lane;
    float v = (float)xr[c];
    xv[t] = v; ss += v * v;
#pragma unroll
    for (int h = 0; h < 4; ++h) {
      qa[h] += v * W[h * WROW + c];
      ka[h] += v * W[h * WROW + Cn + c];
    }
  }
  ss = wred(ss);
#pragma unroll
  for (int h = 0; h < 4; ++h) { qa[h] = wred(qa[h]); ka[h] = wred(ka[h]); }
  float sc = 1.f / fmaxf(sqrtf(ss), 1e-8f);
#pragma unroll
  for (int t = 0; t < 8; ++t) {
    int c = t * 32 + lane;
    xnh[(size_t)rowg * Cn + c] = (_Float16)(xv[t] * sc);
  }
  if (lane < 4) {
    int h = lane;
    float s0 = spat[rowg * 2], s1 = spat[rowg * 2 + 1];
    qt[rowg * 4 + h] = qa[h] + s0 * W[h * WROW + 2*Cn]     + s1 * W[h * WROW + 2*Cn + 1];
    kt[rowg * 4 + h] = ka[h] + s0 * W[h * WROW + 2*Cn + 2] + s1 * W[h * WROW + 2*Cn + 3];
  }
}

// ---------------------------------------------------------------------------
// gram_topk: scores[i0..i0+15][:] = relu(xn_i . xn_j) via WMMA (K=C=256),
// then top-16 per row scattered into the shared present[N] vector.
// 512 threads = 16 waves: WMMA phase wave -> 4 j-tiles; topk phase wave -> row.
// ---------------------------------------------------------------------------
__global__ __launch_bounds__(512) void gram_topk_kernel(
    const _Float16* __restrict__ xn, float* __restrict__ present) {
  __shared__ float sc[16][Nn];                      // 64 KB (LDS, 320 KB avail)
  int b = blockIdx.y, i0 = blockIdx.x * 16;
  int wave = threadIdx.x >> 5, lane = threadIdx.x & 31;
  const _Float16* Xb   = xn + (size_t)b * Nn * Cn;
  const _Float16* Arow = Xb + (size_t)i0 * Cn;
#pragma unroll 1
  for (int t = 0; t < 4; ++t) {
    int j0 = (wave * 4 + t) * 16;
    const _Float16* Brow = Xb + (size_t)j0 * Cn;
    v8f acc = {};
#pragma unroll
    for (int k0 = 0; k0 < Cn; k0 += 32)
      acc = wmma16(load_frag(Arow, Cn, k0), load_frag(Brow, Cn, k0), acc);
    int mhi = (lane >> 4) << 3, n = lane & 15;
#pragma unroll
    for (int r = 0; r < 8; ++r) {
      float v = acc[r];
      sc[r + mhi][j0 + n] = v > 0.f ? v : 0.f;
    }
  }
  __syncthreads();
  float* R = sc[wave];                              // wave handles row `wave`
#pragma unroll 1
  for (int it = 0; it < Kt; ++it) {
    float bv = -1.f; int bi = 0;
#pragma unroll
    for (int j = lane; j < Nn; j += 32) {
      float v = R[j];
      if (v > bv) { bv = v; bi = j; }
    }
#pragma unroll
    for (int m = 16; m >= 1; m >>= 1) {
      float ov = __shfl_xor(bv, m, 32);
      int   oi = __shfl_xor(bi, m, 32);
      if (ov > bv || (ov == bv && oi < bi)) { bv = ov; bi = oi; }
    }
    if (lane == 0) { present[bi] = 1.0f; R[bi] = -1.0f; }
    __builtin_amdgcn_wave_barrier();
  }
}

// ---------------------------------------------------------------------------
// gconv: out[b, g*64+o, n] = relu(sum_i w[g,o,i] * x[b,n,g*64+i] + bias)
// 128 threads = 4 waves, wave -> o-tile; K=64 -> 2 WMMAs.
// ---------------------------------------------------------------------------
__global__ __launch_bounds__(128) void gconv_kernel(
    const _Float16* __restrict__ xh,   // [B][N][C] row-major (B^T rows = n)
    const _Float16* __restrict__ wh,   // [G][64][64]
    const float* __restrict__ bias,
    float* __restrict__ outf,          // [B][C][N]
    _Float16* __restrict__ outh) {     // [B][C][N]
  int b = blockIdx.z, g = blockIdx.y, n0 = blockIdx.x * 16;
  int wave = threadIdx.x >> 5, lane = threadIdx.x & 31;
  int o0 = wave * 16;
  const _Float16* A  = wh + ((size_t)(g * 64 + o0)) * 64;
  const _Float16* Bt = xh + ((size_t)(b * Nn + n0)) * Cn + g * 64;
  v8f acc = {};
  acc = wmma16(load_frag(A, 64, 0),  load_frag(Bt, Cn, 0),  acc);
  acc = wmma16(load_frag(A, 64, 32), load_frag(Bt, Cn, 32), acc);
  int mhi = (lane >> 4) << 3, n = lane & 15;
#pragma unroll
  for (int r = 0; r < 8; ++r) {
    int c = g * 64 + o0 + r + mhi;
    float v = acc[r] + bias[c];
    v = v > 0.f ? v : 0.f;
    size_t idx = ((size_t)b * Cn + c) * Nn + (n0 + n);
    outf[idx] = v;
    outh[idx] = (_Float16)v;
  }
}

// ---------------------------------------------------------------------------
// attn_apply: out[d,i] += sum_j pre[d,j] * A[i,j,h], A built on the fly in a
// double-buffered LDS tile (one barrier per 32-j chunk; the sigmoid VALU work
// for chunk c+1 overlaps the WMMA of chunk c):
//   A = (sigmoid(qt_j + kt_i + bh) * roi[b,i,j] * smask[b,j] * present[j]
//        + (i==j)*(smask[b,i]==0)) / 4
// Per-j terms qt_j+bh and smask_j*present_j/4 staged once per block in LDS.
// 128 threads = 4 waves (wave -> d-tile); A-tile shared by all waves.
// mode 1: write f16 transposed (pre_f + acc) -> out_t_h[b][i][c]
// mode 2: write f32 acc -> out_nc[b][i][c]
// ---------------------------------------------------------------------------
__global__ __launch_bounds__(128) void attn_apply_kernel(
    const _Float16* __restrict__ preh,   // [B][C][N]  (A operand rows c)
    const float* __restrict__ pref,      // [B][C][N]  (mode 1 residual)
    const float* __restrict__ qt,        // [B][N][4]
    const float* __restrict__ ktp,       // [B][N][4]
    const float* __restrict__ bvec,      // [4]
    const float* __restrict__ roi,       // [B][N][N]
    const float* __restrict__ smask,     // [B][N]
    const float* __restrict__ present,   // [N]
    _Float16* __restrict__ out_t_h,      // mode 1: [B][N][C]
    float* __restrict__ out_nc,          // mode 2: [B][N][C]
    int mode) {
  __shared__ __align__(16) _Float16 At[2][16][32];
  __shared__ float qjv[Nn];              // qt[b,j,h] + b[h]
  __shared__ float mj[Nn];               // smask[b,j] * present[j] * 0.25
  __shared__ float ktv[16], fdg[16];
  int b = blockIdx.z, h = blockIdx.y, i0 = blockIdx.x * 16;
  int wave = threadIdx.x >> 5, lane = threadIdx.x & 31;
  int d0 = wave * 16;
  float bh = bvec[h];
  for (int j = threadIdx.x; j < Nn; j += 128) {
    qjv[j] = qt[((size_t)b * Nn + j) * 4 + h] + bh;
    mj[j]  = smask[(size_t)b * Nn + j] * present[j] * 0.25f;
  }
  if (threadIdx.x < 16) {
    int i = i0 + threadIdx.x;
    ktv[threadIdx.x] = ktp[((size_t)b * Nn + i) * 4 + h];
    fdg[threadIdx.x] = (smask[(size_t)b * Nn + i] == 0.f) ? 0.25f : 0.f;
  }
  __syncthreads();
  int ii = threadIdx.x >> 3;           // 0..15
  int jb = (threadIdx.x & 7) * 4;      // 0,4,...,28
  int ig = i0 + ii;
  float kti = ktv[ii], fdi = fdg[ii];
  const float* rrow = roi + ((size_t)b * Nn + ig) * Nn;
  const _Float16* Arows = preh + ((size_t)b * Cn + h * 64 + d0) * Nn;

  auto build = [&](int buf, int j0) {
    float4 rv = *(const float4*)(rrow + j0 + jb);
    float rvv[4] = {rv.x, rv.y, rv.z, rv.w};
#pragma unroll
    for (int t = 0; t < 4; ++t) {
      int j = j0 + jb + t;
      float lg = qjv[j] + kti;
      float s  = 1.f / (1.f + __expf(-lg));
      float v  = s * rvv[t] * mj[j];
      if (j == ig) v += fdi;
      At[buf][ii][jb + t] = (_Float16)v;
    }
  };

  v8f acc = {};
  build(0, 0);
  __syncthreads();
#pragma unroll 1
  for (int c = 0; c < Nn / 32; ++c) {
    int j0 = c * 32;
    acc = wmma16(load_frag(Arows, Nn, j0),
                 load_frag(&At[c & 1][0][0], 32, 0), acc);
    if (c + 1 < Nn / 32) build((c + 1) & 1, j0 + 32);
    __syncthreads();
  }
  int mhi = (lane >> 4) << 3, n = lane & 15;
  int i = i0 + n;
#pragma unroll
  for (int r = 0; r < 8; ++r) {
    int cch = h * 64 + d0 + r + mhi;
    float v = acc[r];
    if (mode == 1) {
      v += pref[((size_t)b * Cn + cch) * Nn + i];
      out_t_h[((size_t)b * Nn + i) * Cn + cch] = (_Float16)v;
    } else {
      out_nc[((size_t)b * Nn + i) * Cn + cch] = v;
    }
  }
}

// ---------------------------------------------------------------------------
// final: per (b,n): LayerNorm(out2m[b,n,:]) * g + b, + out2_pre[b,:,n]
// ---------------------------------------------------------------------------
__global__ __launch_bounds__(256) void final_ln_kernel(
    const float* __restrict__ o2m,   // [B][N][C]
    const float* __restrict__ pre,   // [B][C][N]
    const float* __restrict__ lg, const float* __restrict__ lb,
    float* __restrict__ out) {       // [B][N][C]
  int wave = threadIdx.x >> 5, lane = threadIdx.x & 31;
  int rowg = blockIdx.x * 8 + wave;
  int b = rowg >> 10, nn = rowg & (Nn - 1);
  const float* r = o2m + (size_t)rowg * Cn;
  float xv[8], s = 0.f;
#pragma unroll
  for (int t = 0; t < 8; ++t) { int c = t * 32 + lane; xv[t] = r[c]; s += xv[t]; }
  s = wred(s);
  float mu = s * (1.f / Cn), vs = 0.f;
#pragma unroll
  for (int t = 0; t < 8; ++t) { float d = xv[t] - mu; vs += d * d; }
  vs = wred(vs);
  float inv = rsqrtf(vs * (1.f / Cn) + 1e-6f);
#pragma unroll
  for (int t = 0; t < 8; ++t) {
    int c = t * 32 + lane;
    float ln = (xv[t] - mu) * inv * lg[c] + lb[c];
    out[(size_t)rowg * Cn + c] = pre[((size_t)b * Cn + c) * Nn + nn] + ln;
  }
}

// ---------------------------------------------------------------------------
extern "C" void kernel_launch(void* const* d_in, const int* in_sizes, int n_in,
                              void* d_out, int out_size, void* d_ws, size_t ws_size,
                              hipStream_t stream) {
  const float* input = (const float*)d_in[0];
  const float* roi   = (const float*)d_in[1];
  const float* smask = (const float*)d_in[2];
  const float* spat  = (const float*)d_in[3];
  const float* W1    = (const float*)d_in[4];
  const float* b1    = (const float*)d_in[5];
  const float* W2    = (const float*)d_in[6];
  const float* b2    = (const float*)d_in[7];
  const float* cw1   = (const float*)d_in[8];
  const float* cb1   = (const float*)d_in[9];
  const float* cw2   = (const float*)d_in[10];
  const float* cb2   = (const float*)d_in[11];
  const float* lng   = (const float*)d_in[12];
  const float* lnb   = (const float*)d_in[13];
  float* out = (float*)d_out;

  char* ws = (char*)d_ws;
  size_t off = 0;
  auto alloc = [&](size_t bytes) -> void* {
    void* p = ws + off;
    off += (bytes + 255) & ~(size_t)255;
    return p;
  };
  const size_t BNC = (size_t)Bn * Nn * Cn;
  _Float16* x1h  = (_Float16*)alloc(BNC * 2);  // input f16      [B][N][C]
  _Float16* xn1  = (_Float16*)alloc(BNC * 2);  // normalized x1  [B][N][C]
  _Float16* xn2  = (_Float16*)alloc(BNC * 2);  // normalized x2  [B][N][C]
  _Float16* o1th = (_Float16*)alloc(BNC * 2);  // out1 (post-res)[B][N][C]
  _Float16* o1ph = (_Float16*)alloc(BNC * 2);  // out1_pre f16   [B][C][N]
  _Float16* o2ph = (_Float16*)alloc(BNC * 2);  // out2_pre f16   [B][C][N]
  float* o1pf = (float*)alloc(BNC * 4);        // out1_pre f32   [B][C][N]
  float* o2pf = (float*)alloc(BNC * 4);        // out2_pre f32   [B][C][N]
  float* o2m  = (float*)alloc(BNC * 4);        // out2m          [B][N][C]
  float* qt1 = (float*)alloc((size_t)Bn * Nn * 4 * 4);
  float* kt1 = (float*)alloc((size_t)Bn * Nn * 4 * 4);
  float* qt2 = (float*)alloc((size_t)Bn * Nn * 4 * 4);
  float* kt2 = (float*)alloc((size_t)Bn * Nn * 4 * 4);
  float* p1  = (float*)alloc(Nn * 4);
  float* p2  = (float*)alloc(Nn * 4);
  _Float16* w1h = (_Float16*)alloc((size_t)Hn * 64 * 64 * 2);
  _Float16* w2h = (_Float16*)alloc((size_t)Hn * 64 * 64 * 2);

  misc_kernel<<<1, 256, 0, stream>>>(cw1, cw2, w1h, w2h, p1, p2);
  prep1_kernel<<<Bn * Nn / 8, 256, 0, stream>>>(input, spat, W1, x1h, xn1, qt1, kt1);
  gram_topk_kernel<<<dim3(Nn / 16, Bn), 512, 0, stream>>>(xn1, p1);
  gconv_kernel<<<dim3(Nn / 16, Hn, Bn), 128, 0, stream>>>(x1h, w1h, cb1, o1pf, o1ph);
  attn_apply_kernel<<<dim3(Nn / 16, Hn, Bn), 128, 0, stream>>>(
      o1ph, o1pf, qt1, kt1, b1, roi, smask, p1, o1th, nullptr, 1);
  prep2_kernel<<<Bn * Nn / 8, 256, 0, stream>>>(o1th, spat, W2, xn2, qt2, kt2);
  gram_topk_kernel<<<dim3(Nn / 16, Bn), 512, 0, stream>>>(xn2, p2);
  gconv_kernel<<<dim3(Nn / 16, Hn, Bn), 128, 0, stream>>>(o1th, w2h, cb2, o2pf, o2ph);
  attn_apply_kernel<<<dim3(Nn / 16, Hn, Bn), 128, 0, stream>>>(
      o2ph, nullptr, qt2, kt2, b2, roi, smask, p2, nullptr, o2m, 2);
  final_ln_kernel<<<Bn * Nn / 8, 256, 0, stream>>>(o2m, o2pf, lng, lnb, out);
}
